// GCU_W_60241211294347
// MI455X (gfx1250) — matrix operand order repaired and checked by reading, not compiled, not run on hardware
//
#include <hip/hip_runtime.h>

// Problem constants (match reference)
#define BB   4
#define CC   256
#define HH   128
#define WW   256
#define CQK  64
#define COUT 256

typedef __attribute__((ext_vector_type(16))) __bf16 v16bf;
typedef __attribute__((ext_vector_type(8)))  float  v8f;
typedef __attribute__((ext_vector_type(2)))  float  v2f;
typedef __attribute__((ext_vector_type(2)))  __bf16 v2bf;

union FragBF { unsigned short u[16]; v16bf v; };   // 32 B, 32-B aligned

__device__ __forceinline__ unsigned short f2bf(float f) {
    __bf16 h = (__bf16)f;                       // RNE, native cvt when available
    return __builtin_bit_cast(unsigned short, h);
}
__device__ __forceinline__ unsigned int packbf2(float a, float b) {
    v2f x = {a, b};
    v2bf y = __builtin_convertvector(x, v2bf);  // v_cvt_pk_bf16_f32 class
    return __builtin_bit_cast(unsigned int, y);
}

// LDS layout (bytes): total 320 KB
//   [0      ,131072) Xcm  bf16[256 w][256 c]  staging (col-major) -> later f32 energy+bf16 att
//   [131072 ,262144) Vu   bf16[256 o][256 w']
//   [262144 ,294912) Kcm  bf16[256 w'][64 d]
//   [294912 ,327680) Qrm  bf16[256 wq][64 d]
#define SMEM_BYTES 327680

__global__ void __launch_bounds__(256)
axial_attn_wmma_kernel(const float* __restrict__ flow,
                       const float* __restrict__ de_out,
                       const float* __restrict__ Wq, const float* __restrict__ bq,
                       const float* __restrict__ Wk, const float* __restrict__ bk,
                       const float* __restrict__ Wv, const float* __restrict__ bv,
                       float* __restrict__ out)
{
    extern __shared__ char smem[];
    unsigned short* Xcm = (unsigned short*)smem;
    float*          Ef  = (float*)smem;                       // aliases Xcm in phase 4
    unsigned short* Vu  = (unsigned short*)(smem + 131072);
    unsigned short* Kcm = (unsigned short*)(smem + 262144);
    unsigned short* Qrm = (unsigned short*)(smem + 294912);

    const int tid  = threadIdx.x;
    const int lane = tid & 31;
    const int wave = tid >> 5;
    const int g    = lane >> 4;   // half-wave group
    const int lm   = lane & 15;

    const int b = blockIdx.x / HH;
    const int h = blockIdx.x % HH;
    const size_t chanStride = (size_t)HH * WW;

    // ---- stage helper: global f32 slice -> LDS bf16 column-major ---------
    auto stage = [&](const float* __restrict__ src) {
        for (int i4 = tid; i4 < (CC * WW) / 4; i4 += 256) {
            int c = i4 >> 6, w4 = i4 & 63;
            float4 v = ((const float4*)(src + (size_t)c * chanStride))[w4];
            int w = w4 * 4;
            Xcm[(w + 0) * CC + c] = f2bf(v.x);
            Xcm[(w + 1) * CC + c] = f2bf(v.y);
            Xcm[(w + 2) * CC + c] = f2bf(v.z);
            Xcm[(w + 3) * CC + c] = f2bf(v.w);
        }
    };

    // A-fragment from global f32 row-major weights (two runs of 8, bf16-packed)
    auto loadWA = [&](FragBF& f, const float* __restrict__ wrow, int k0) {
        const float4* p0 = (const float4*)(wrow + k0 + 8 * g);
        float4 a0 = p0[0], a1 = p0[1];
        const float4* p1 = (const float4*)(wrow + k0 + 16 + 8 * g);
        float4 c0 = p1[0], c1 = p1[1];
        unsigned int* q = (unsigned int*)f.u;
        q[0] = packbf2(a0.x, a0.y); q[1] = packbf2(a0.z, a0.w);
        q[2] = packbf2(a1.x, a1.y); q[3] = packbf2(a1.z, a1.w);
        q[4] = packbf2(c0.x, c0.y); q[5] = packbf2(c0.z, c0.w);
        q[6] = packbf2(c1.x, c1.y); q[7] = packbf2(c1.z, c1.w);
    };
    // A-fragment from LDS row-major bf16 (two ds_load_b128)
    auto ldsA = [&](FragBF& f, const unsigned short* row, int k0) {
        *(uint4*)&f.u[0] = *(const uint4*)(row + k0 + 8 * g);
        *(uint4*)&f.u[8] = *(const uint4*)(row + k0 + 16 + 8 * g);
    };
    // B-fragment from LDS column-major bf16 (two 16-byte runs, table k order)
    auto ldsB = [&](FragBF& f, const unsigned short* col, int k0) {
        *(uint4*)&f.u[0] = *(const uint4*)(col + k0 + 8 * g);
        *(uint4*)&f.u[8] = *(const uint4*)(col + k0 + 16 + 8 * g);
    };

    // ---- Phase 1: stage flow[b,:,h,:] ------------------------------------
    stage(flow + (size_t)b * CC * chanStride + (size_t)h * WW);
    __syncthreads();

    // ---- Phase 2a: K = Wk*Xf + bk -> Kcm[w'][d] --------------------------
    // 4 row-panels x 2 half-N-ranges == 8 waves exactly (uniform, no loop)
    {
        const int mt = wave >> 1;
        const int n0 = (wave & 1) * 8;
        const float* wrow = Wk + (size_t)(mt * 16 + lm) * CC;
        FragBF af[8];
#pragma unroll
        for (int kk = 0; kk < 8; ++kk) loadWA(af[kk], wrow, kk * 32);
        const float4* bp = (const float4*)(bk + mt * 16 + 8 * g);
        float4 bl = bp[0], bh = bp[1];
        float bv8[8] = {bl.x, bl.y, bl.z, bl.w, bh.x, bh.y, bh.z, bh.w};
        for (int nt = n0; nt < n0 + 8; ++nt) {
            const unsigned short* colp = Xcm + (nt * 16 + lm) * CC;
            FragBF bf8[8];
#pragma unroll
            for (int kk = 0; kk < 8; ++kk) ldsB(bf8[kk], colp, kk * 32);
            v8f acc = {};
#pragma unroll
            for (int kk = 0; kk < 8; ++kk)
                acc = __builtin_amdgcn_wmma_f32_16x16x32_bf16(
                          false, af[kk].v, false, bf8[kk].v, (short)0, acc, false, false);
            uint4 pk;
            pk.x = packbf2(acc[0] + bv8[0], acc[1] + bv8[1]);
            pk.y = packbf2(acc[2] + bv8[2], acc[3] + bv8[3]);
            pk.z = packbf2(acc[4] + bv8[4], acc[5] + bv8[5]);
            pk.w = packbf2(acc[6] + bv8[6], acc[7] + bv8[7]);
            *(uint4*)(Kcm + (nt * 16 + lm) * CQK + mt * 16 + 8 * g) = pk;
        }
    }

    // ---- Phase 2b: V = Wv*Xf + bv -> Vu[o][w'] ---------------------------
    for (int p = wave; p < 16; p += 8) {
        const int mt = p;
        const float* wrow = Wv + (size_t)(mt * 16 + lm) * CC;
        FragBF af[8];
#pragma unroll
        for (int kk = 0; kk < 8; ++kk) loadWA(af[kk], wrow, kk * 32);
        const float4* bp = (const float4*)(bv + mt * 16 + 8 * g);
        float4 bl = bp[0], bh = bp[1];
        float bv8[8] = {bl.x, bl.y, bl.z, bl.w, bh.x, bh.y, bh.z, bh.w};
        for (int nt = 0; nt < 16; ++nt) {
            const unsigned short* colp = Xcm + (nt * 16 + lm) * CC;
            FragBF bf8[8];
#pragma unroll
            for (int kk = 0; kk < 8; ++kk) ldsB(bf8[kk], colp, kk * 32);
            v8f acc = {};
#pragma unroll
            for (int kk = 0; kk < 8; ++kk)
                acc = __builtin_amdgcn_wmma_f32_16x16x32_bf16(
                          false, af[kk].v, false, bf8[kk].v, (short)0, acc, false, false);
#pragma unroll
            for (int r = 0; r < 8; ++r)
                Vu[(mt * 16 + 8 * g + r) * WW + nt * 16 + lm] = f2bf(acc[r] + bv8[r]);
        }
    }
    __syncthreads();

    // ---- Phase 3: stage de_out, compute Q = Wq*Xde + bq -> Qrm[wq][d] ----
    stage(de_out + (size_t)b * CC * chanStride + (size_t)h * WW);
    __syncthreads();
    {
        const int mt = wave >> 1;
        const int n0 = (wave & 1) * 8;
        const float* wrow = Wq + (size_t)(mt * 16 + lm) * CC;
        FragBF af[8];
#pragma unroll
        for (int kk = 0; kk < 8; ++kk) loadWA(af[kk], wrow, kk * 32);
        const float4* bp = (const float4*)(bq + mt * 16 + 8 * g);
        float4 bl = bp[0], bh = bp[1];
        float bv8[8] = {bl.x, bl.y, bl.z, bl.w, bh.x, bh.y, bh.z, bh.w};
        for (int nt = n0; nt < n0 + 8; ++nt) {
            const unsigned short* colp = Xcm + (nt * 16 + lm) * CC;
            FragBF bf8[8];
#pragma unroll
            for (int kk = 0; kk < 8; ++kk) ldsB(bf8[kk], colp, kk * 32);
            v8f acc = {};
#pragma unroll
            for (int kk = 0; kk < 8; ++kk)
                acc = __builtin_amdgcn_wmma_f32_16x16x32_bf16(
                          false, af[kk].v, false, bf8[kk].v, (short)0, acc, false, false);
            uint4 pk;
            pk.x = packbf2(acc[0] + bv8[0], acc[1] + bv8[1]);
            pk.y = packbf2(acc[2] + bv8[2], acc[3] + bv8[3]);
            pk.z = packbf2(acc[4] + bv8[4], acc[5] + bv8[5]);
            pk.w = packbf2(acc[6] + bv8[6], acc[7] + bv8[7]);
            *(uint4*)(Qrm + (nt * 16 + lm) * CQK + mt * 16 + 8 * g) = pk;
        }
    }
    __syncthreads();

    // ---- Phase 4: attention in two 128-query-row blocks ------------------
    for (int wblk = 0; wblk < 2; ++wblk) {
        const int wqbase = wblk * 128;

        // 4a: E[wq, w'] = sum_d Q[wq,d] * K[w',d]   (K-dim = 64); 1 panel/wave
        {
            const int mt = wave;
            const unsigned short* qrow = Qrm + (wqbase + mt * 16 + lm) * CQK;
            FragBF af[2];
#pragma unroll
            for (int kk = 0; kk < 2; ++kk) ldsA(af[kk], qrow, kk * 32);
            for (int nt = 0; nt < 16; ++nt) {
                const unsigned short* colp = Kcm + (nt * 16 + lm) * CQK;
                FragBF bf2[2];
#pragma unroll
                for (int kk = 0; kk < 2; ++kk) ldsB(bf2[kk], colp, kk * 32);
                v8f acc = {};
#pragma unroll
                for (int kk = 0; kk < 2; ++kk)
                    acc = __builtin_amdgcn_wmma_f32_16x16x32_bf16(
                              false, af[kk].v, false, bf2[kk].v, (short)0, acc, false, false);
#pragma unroll
                for (int r = 0; r < 8; ++r)
                    Ef[(mt * 16 + 8 * g + r) * WW + nt * 16 + lm] = acc[r];
            }
        }
        __syncthreads();

        // 4b: wave-parallel row softmax: one wave per row, 8 cols per lane
        for (int row = wave; row < 128; row += 8) {
            float* Er = Ef + row * WW;
            const float4* Ep = (const float4*)Er;
            float4 x0 = Ep[lane * 2], x1 = Ep[lane * 2 + 1];
            float mx = fmaxf(fmaxf(fmaxf(x0.x, x0.y), fmaxf(x0.z, x0.w)),
                             fmaxf(fmaxf(x1.x, x1.y), fmaxf(x1.z, x1.w)));
#pragma unroll
            for (int m = 16; m >= 1; m >>= 1) mx = fmaxf(mx, __shfl_xor(mx, m, 32));
            x0.x = __expf(x0.x - mx); x0.y = __expf(x0.y - mx);
            x0.z = __expf(x0.z - mx); x0.w = __expf(x0.w - mx);
            x1.x = __expf(x1.x - mx); x1.y = __expf(x1.y - mx);
            x1.z = __expf(x1.z - mx); x1.w = __expf(x1.w - mx);
            float sum = (x0.x + x0.y) + (x0.z + x0.w) + (x1.x + x1.y) + (x1.z + x1.w);
#pragma unroll
            for (int m = 16; m >= 1; m >>= 1) sum += __shfl_xor(sum, m, 32);
            float inv = 1.f / sum;
            uint4 pk;                               // bf16 att row packed in place
            pk.x = packbf2(x0.x * inv, x0.y * inv);
            pk.y = packbf2(x0.z * inv, x0.w * inv);
            pk.z = packbf2(x1.x * inv, x1.y * inv);
            pk.w = packbf2(x1.z * inv, x1.w * inv);
            *(uint4*)((unsigned short*)Er + lane * 8) = pk;
        }
        __syncthreads();

        // 4c: out^T tiles: D[M=wq, N=o] = sum_{w'} att[wq,w'] * V[o,w']
        {
            unsigned short* attu = (unsigned short*)Ef;     // row stride = 512 u16
            const int mt = wave;
            const unsigned short* arow = attu + (mt * 16 + lm) * 512;
            FragBF af[8];
#pragma unroll
            for (int kk = 0; kk < 8; ++kk) ldsA(af[kk], arow, kk * 32);
            for (int nt = 0; nt < 16; ++nt) {
                const unsigned short* colp = Vu + (nt * 16 + lm) * WW;
                FragBF bf8[8];
#pragma unroll
                for (int kk = 0; kk < 8; ++kk) ldsB(bf8[kk], colp, kk * 32);
                v8f acc = {};
#pragma unroll
                for (int kk = 0; kk < 8; ++kk)
                    acc = __builtin_amdgcn_wmma_f32_16x16x32_bf16(
                              false, af[kk].v, false, bf8[kk].v, (short)0, acc, false, false);
                // lane holds 8 consecutive wq for output channel o = nt*16+lm
                float* op = out + (((size_t)b * COUT + nt * 16 + lm) * HH + h) * WW
                                + wqbase + mt * 16 + 8 * g;
                ((float4*)op)[0] = make_float4(acc[0], acc[1], acc[2], acc[3]);
                ((float4*)op)[1] = make_float4(acc[4], acc[5], acc[6], acc[7]);
            }
        }
        __syncthreads();
    }
}

extern "C" void kernel_launch(void* const* d_in, const int* in_sizes, int n_in,
                              void* d_out, int out_size, void* d_ws, size_t ws_size,
                              hipStream_t stream) {
    (void)in_sizes; (void)n_in; (void)out_size; (void)d_ws; (void)ws_size;
    const float* flow   = (const float*)d_in[0];
    const float* de_out = (const float*)d_in[1];
    const float* Wq     = (const float*)d_in[2];
    const float* bq     = (const float*)d_in[3];
    const float* Wk     = (const float*)d_in[4];
    const float* bk     = (const float*)d_in[5];
    const float* Wv     = (const float*)d_in[6];
    const float* bv     = (const float*)d_in[7];
    float* out = (float*)d_out;

    hipFuncSetAttribute((const void*)axial_attn_wmma_kernel,
                        hipFuncAttributeMaxDynamicSharedMemorySize, SMEM_BYTES);

    dim3 grid(BB * HH);   // one workgroup per (b, h) row
    dim3 block(256);      // 8 wave32s
    axial_attn_wmma_kernel<<<grid, block, SMEM_BYTES, stream>>>(
        flow, de_out, Wq, bq, Wk, bk, Wv, bv, out);
}
